// Seq2Seq_3539053052489
// MI455X (gfx1250) — matrix operand order repaired and checked by reading, not compiled, not run on hardware
//
#include <hip/hip_runtime.h>
#include <hip/hip_bf16.h>

// Problem constants (fixed by the reference harness)
#define Vv 512
#define Ee 128
#define Hh 256
#define Bb 2048
#define Ss 32
#define Tt 48
#define KC (Ee + Hh)   // 384, concatenated K for [x,h] @ [Wih;Whh]^T

typedef __bf16 bf16_t;
typedef __attribute__((ext_vector_type(16))) __bf16 v16bf;
typedef __attribute__((ext_vector_type(8)))  __bf16 v8bf;
typedef __attribute__((ext_vector_type(8)))  float  v8f;

// ---------------- helpers ----------------

__device__ __forceinline__ bf16_t f2bf(float f) {
    union { float f; unsigned u; } a; a.f = f;
    unsigned u = a.u;
    unsigned r = u + 0x7fffu + ((u >> 16) & 1u);   // round-to-nearest-even
    union { unsigned short s; bf16_t b; } o;
    o.s = (unsigned short)(r >> 16);
    return o.b;
}

__device__ __forceinline__ float fast_sigmoid(float x) {
    return 1.0f / (1.0f + __expf(-x));
}

__device__ __forceinline__ float fast_tanh(float x) {
    float cx = fminf(fmaxf(x, -10.0f), 10.0f);
    float e = __expf(2.0f * cx);
    return (e - 1.0f) / (e + 1.0f);
}

union AFrag { v16bf v; v8bf h[2]; };

// ---------------- single fused prep kernel ----------------
// Segmented index dispatch over all one-time setup work:
//  [0]            enc Wcat bf16 [4H][KC]
//  [1]            dec Wcat bf16 [4H][KC]
//  [2],[3]        enc/dec bias sums [4H] f32
//  [4]            fc_W -> bf16 [V][H]
//  [5],[6]        enc/dec embedding -> bf16 [V][E]
//  [7]            h0 = 0 (bf16), c = 0 (f32), tok = 0
__global__ __launch_bounds__(256) void k_prep(
    const float* __restrict__ encWih, const float* __restrict__ encWhh,
    const float* __restrict__ encbih, const float* __restrict__ encbhh,
    const float* __restrict__ decWih, const float* __restrict__ decWhh,
    const float* __restrict__ decbih, const float* __restrict__ decbhh,
    const float* __restrict__ fcW,    const float* __restrict__ encEmb,
    const float* __restrict__ decEmb,
    bf16_t* __restrict__ encW,  bf16_t* __restrict__ decW,
    float*  __restrict__ encb,  float*  __restrict__ decb,
    bf16_t* __restrict__ fcWb,  bf16_t* __restrict__ encEmbB,
    bf16_t* __restrict__ decEmbB,
    bf16_t* __restrict__ h0,    float* __restrict__ c, int* __restrict__ tok)
{
    int idx = blockIdx.x * 256 + threadIdx.x;
    const int SZW = 4 * Hh * KC;   // 393216
    const int SZB = 4 * Hh;        // 1024
    const int SZF = Vv * Hh;       // 131072
    const int SZE = Vv * Ee;       // 65536
    const int SZH = Bb * Hh;       // 524288

    if (idx < SZW) {
        int n = idx / KC, k = idx % KC;
        encW[idx] = f2bf(k < Ee ? encWih[n * Ee + k] : encWhh[n * Hh + (k - Ee)]);
        return;
    }
    idx -= SZW;
    if (idx < SZW) {
        int n = idx / KC, k = idx % KC;
        decW[idx] = f2bf(k < Ee ? decWih[n * Ee + k] : decWhh[n * Hh + (k - Ee)]);
        return;
    }
    idx -= SZW;
    if (idx < SZB) { encb[idx] = encbih[idx] + encbhh[idx]; return; }
    idx -= SZB;
    if (idx < SZB) { decb[idx] = decbih[idx] + decbhh[idx]; return; }
    idx -= SZB;
    if (idx < SZF) { fcWb[idx] = f2bf(fcW[idx]); return; }
    idx -= SZF;
    if (idx < SZE) { encEmbB[idx] = f2bf(encEmb[idx]); return; }
    idx -= SZE;
    if (idx < SZE) { decEmbB[idx] = f2bf(decEmb[idx]); return; }
    idx -= SZE;
    if (idx < SZH) {
        union { unsigned short s; bf16_t b; } z; z.s = 0;
        h0[idx] = z.b;
        c[idx] = 0.0f;
        if (idx < Bb) tok[idx] = 0;
        return;
    }
}

// ---------------- fused embed + LSTM step (WMMA bf16) ----------------
// grid = (B/32, H/64), block = 256 (8 waves).
// wave w: rh = w>>2 picks 16-row half, cq = w&3 picks 16-col quarter of 64-col j-tile.
// Each wave holds 4 accumulators: gate strips i,f,g,o at columns g*H + j.
// The x_t input is gathered on the fly from the bf16 embedding table:
//   token = tokidx[row*tokstride + tokoff]  (encoder: src[b*S+t]; decoder: tok[b])
// Inner loop loads A + all 4 B fragments first, then issues 4 WMMAs, so the
// compiler can clause the loads and overlap WMMA with the next loads.
__global__ __launch_bounds__(256) void k_lstm_step(
    const bf16_t* __restrict__ Emb,    // [V,E] bf16 embedding table
    const int*    __restrict__ tokidx, // token indices
    int tokstride, int tokoff,
    const bf16_t* __restrict__ Hin,    // [B,H]
    bf16_t* __restrict__ Hout,         // [B,H]
    float* __restrict__ C,             // [B,H]
    const bf16_t* __restrict__ W,      // [4H,KC]
    const float* __restrict__ bias)    // [4H]
{
    const int lane = threadIdx.x & 31;
    const int wave = threadIdx.x >> 5;
    const int rh   = wave >> 2;
    const int cq   = wave & 3;
    const int row0 = blockIdx.x * 32 + rh * 16;
    const int j0   = blockIdx.y * 64 + cq * 16;
    const int hsel = lane >> 4;     // which half of the wave
    const int lid  = lane & 15;

    // Per-lane A-matrix row pointers (lane handles row row0+lid of A)
    const int arow = row0 + lid;
    const int token = tokidx[arow * tokstride + tokoff];
    const bf16_t* xrow = Emb + (size_t)token * Ee;
    const bf16_t* hrow = Hin + (size_t)arow * Hh;
    // Per-lane base for the four gate-strip weight rows
    const bf16_t* wbase = W + (size_t)(j0 + lid) * KC + hsel * 16;

    v8f acc0 = (v8f)0.0f, acc1 = (v8f)0.0f, acc2 = (v8f)0.0f, acc3 = (v8f)0.0f;

#pragma unroll
    for (int kk = 0; kk < KC / 32; ++kk) {
        // A fragment: lane (hsel,lid): K chunks [kb+hsel*16, +8) and +8
        const bf16_t* ap = (kk < Ee / 32)
                             ? (xrow + kk * 32 + hsel * 16)
                             : (hrow + (kk - Ee / 32) * 32 + hsel * 16);
        AFrag a;
        a.h[0] = *(const v8bf*)(ap);
        a.h[1] = *(const v8bf*)(ap + 8);

        // Load all 4 gate B fragments up front (lane holds col g*H + j0 + lid,
        // 16 contiguous K values), then issue the 4 WMMAs back-to-back.
        v16bf b0 = *(const v16bf*)(wbase + (size_t)(0 * Hh) * KC + kk * 32);
        v16bf b1 = *(const v16bf*)(wbase + (size_t)(1 * Hh) * KC + kk * 32);
        v16bf b2 = *(const v16bf*)(wbase + (size_t)(2 * Hh) * KC + kk * 32);
        v16bf b3 = *(const v16bf*)(wbase + (size_t)(3 * Hh) * KC + kk * 32);

        acc0 = __builtin_amdgcn_wmma_f32_16x16x32_bf16(false, a.v, false, b0, (short)0, acc0, false, false);
        acc1 = __builtin_amdgcn_wmma_f32_16x16x32_bf16(false, a.v, false, b1, (short)0, acc1, false, false);
        acc2 = __builtin_amdgcn_wmma_f32_16x16x32_bf16(false, a.v, false, b2, (short)0, acc2, false, false);
        acc3 = __builtin_amdgcn_wmma_f32_16x16x32_bf16(false, a.v, false, b3, (short)0, acc3, false, false);
    }

    // Fused LSTM nonlinearity on accumulator registers.
    // C/D layout: VGPR r, lane -> M = r + hsel*8, N = lid
    const int j = j0 + lid;
    const float bi = bias[0 * Hh + j];
    const float bf = bias[1 * Hh + j];
    const float bg = bias[2 * Hh + j];
    const float bo = bias[3 * Hh + j];

#pragma unroll
    for (int r = 0; r < 8; ++r) {
        const int m = row0 + hsel * 8 + r;
        float iv = fast_sigmoid(acc0[r] + bi);
        float fv = fast_sigmoid(acc1[r] + bf);
        float gv = fast_tanh(acc2[r] + bg);
        float ov = fast_sigmoid(acc3[r] + bo);
        float cold = C[m * Hh + j];
        float cn = fv * cold + iv * gv;
        float hn = ov * fast_tanh(cn);
        C[m * Hh + j] = cn;
        Hout[m * Hh + j] = f2bf(hn);
    }
}

// ---------------- logits GEMM + bias + argmax (WMMA bf16) ----------------
// grid = B/16, block = 256. Block computes 16 rows x full V=512 cols.
// wave w covers columns [w*64, w*64+64) via 4 accumulators.
__global__ __launch_bounds__(256) void k_logits_argmax(
    const bf16_t* __restrict__ Hbf, // [B,H]
    const bf16_t* __restrict__ Wfc, // [V,H] bf16
    const float* __restrict__ bfc,  // [V]
    float* __restrict__ outp,       // [B,T,V]
    int* __restrict__ tok,          // [B]
    int t)
{
    __shared__ float sL[16 * Vv];
    __shared__ float sBv[16 * 16];
    __shared__ int   sBi[16 * 16];

    const int lane = threadIdx.x & 31;
    const int wave = threadIdx.x >> 5;
    const int row0 = blockIdx.x * 16;
    const int hsel = lane >> 4;
    const int lid  = lane & 15;

    const bf16_t* hrow  = Hbf + (size_t)(row0 + lid) * Hh + hsel * 16;
    const bf16_t* wbase = Wfc + (size_t)(wave * 64 + lid) * Hh + hsel * 16;

    v8f acc0 = (v8f)0.0f, acc1 = (v8f)0.0f, acc2 = (v8f)0.0f, acc3 = (v8f)0.0f;

#pragma unroll
    for (int kk = 0; kk < Hh / 32; ++kk) {
        AFrag a;
        a.h[0] = *(const v8bf*)(hrow + kk * 32);
        a.h[1] = *(const v8bf*)(hrow + kk * 32 + 8);

        v16bf b0 = *(const v16bf*)(wbase + (size_t)(0 * 16) * Hh + kk * 32);
        v16bf b1 = *(const v16bf*)(wbase + (size_t)(1 * 16) * Hh + kk * 32);
        v16bf b2 = *(const v16bf*)(wbase + (size_t)(2 * 16) * Hh + kk * 32);
        v16bf b3 = *(const v16bf*)(wbase + (size_t)(3 * 16) * Hh + kk * 32);

        acc0 = __builtin_amdgcn_wmma_f32_16x16x32_bf16(false, a.v, false, b0, (short)0, acc0, false, false);
        acc1 = __builtin_amdgcn_wmma_f32_16x16x32_bf16(false, a.v, false, b1, (short)0, acc1, false, false);
        acc2 = __builtin_amdgcn_wmma_f32_16x16x32_bf16(false, a.v, false, b2, (short)0, acc2, false, false);
        acc3 = __builtin_amdgcn_wmma_f32_16x16x32_bf16(false, a.v, false, b3, (short)0, acc3, false, false);
    }

#pragma unroll
    for (int q = 0; q < 4; ++q) {
        const int n = wave * 64 + q * 16 + lid;
        const float bv = bfc[n];
        const v8f accq = (q == 0) ? acc0 : (q == 1) ? acc1 : (q == 2) ? acc2 : acc3;
#pragma unroll
        for (int r = 0; r < 8; ++r) {
            const int ml = hsel * 8 + r;
            float v = accq[r] + bv;
            outp[(size_t)(row0 + ml) * ((size_t)Tt * Vv) + (size_t)t * Vv + n] = v;
            sL[ml * Vv + n] = v;
        }
    }
    __syncthreads();

    // Two-level argmax over 512 cols per row; first-occurrence tie-break.
    const int row = threadIdx.x >> 4;   // 0..15
    const int seg = threadIdx.x & 15;   // 0..15
    float bvv = -3.4e38f; int bii = 0;
    for (int q = 0; q < 32; ++q) {
        int n = seg * 32 + q;
        float v = sL[row * Vv + n];
        if (v > bvv) { bvv = v; bii = n; }
    }
    sBv[row * 16 + seg] = bvv;
    sBi[row * 16 + seg] = bii;
    __syncthreads();
    if (seg == 0) {
        float best = sBv[row * 16];
        int bidx = sBi[row * 16];
        for (int s2 = 1; s2 < 16; ++s2) {
            float v = sBv[row * 16 + s2];
            if (v > best) { best = v; bidx = sBi[row * 16 + s2]; }
        }
        tok[row0 + row] = bidx;
    }
}

// ---------------- host launcher ----------------

extern "C" void kernel_launch(void* const* d_in, const int* in_sizes, int n_in,
                              void* d_out, int out_size, void* d_ws, size_t ws_size,
                              hipStream_t stream) {
    (void)in_sizes; (void)n_in; (void)out_size; (void)ws_size;

    const int*   src     = (const int*)  d_in[0];
    // d_in[1] = target_len (compile-time Tt)
    const float* enc_emb = (const float*)d_in[2];
    const float* enc_Wih = (const float*)d_in[3];
    const float* enc_Whh = (const float*)d_in[4];
    const float* enc_bih = (const float*)d_in[5];
    const float* enc_bhh = (const float*)d_in[6];
    const float* dec_emb = (const float*)d_in[7];
    const float* dec_Wih = (const float*)d_in[8];
    const float* dec_Whh = (const float*)d_in[9];
    const float* dec_bih = (const float*)d_in[10];
    const float* dec_bhh = (const float*)d_in[11];
    const float* fc_W    = (const float*)d_in[12];
    const float* fc_b    = (const float*)d_in[13];
    float* outp = (float*)d_out;

    // workspace carve-out (256B aligned slabs)
    char* ws = (char*)d_ws;
    size_t off = 0;
    auto carve = [&](size_t bytes) -> char* {
        off = (off + 255) & ~(size_t)255;
        char* p = ws + off;
        off += bytes;
        return p;
    };
    bf16_t* encW   = (bf16_t*)carve((size_t)4 * Hh * KC * sizeof(bf16_t));
    bf16_t* decW   = (bf16_t*)carve((size_t)4 * Hh * KC * sizeof(bf16_t));
    float*  encb   = (float*) carve((size_t)4 * Hh * sizeof(float));
    float*  decb   = (float*) carve((size_t)4 * Hh * sizeof(float));
    bf16_t* fcWb   = (bf16_t*)carve((size_t)Vv * Hh * sizeof(bf16_t));
    bf16_t* encEmB = (bf16_t*)carve((size_t)Vv * Ee * sizeof(bf16_t));
    bf16_t* decEmB = (bf16_t*)carve((size_t)Vv * Ee * sizeof(bf16_t));
    bf16_t* hbuf0  = (bf16_t*)carve((size_t)Bb * Hh * sizeof(bf16_t));
    bf16_t* hbuf1  = (bf16_t*)carve((size_t)Bb * Hh * sizeof(bf16_t));
    float*  cbuf   = (float*) carve((size_t)Bb * Hh * sizeof(float));
    int*    tok    = (int*)   carve((size_t)Bb * sizeof(int));

    // ---- one fused prep launch ----
    {
        const int total = 2 * (4 * Hh * KC) + 2 * (4 * Hh) + Vv * Hh + 2 * (Vv * Ee) + Bb * Hh;
        const int nblk = (total + 255) / 256;
        k_prep<<<nblk, 256, 0, stream>>>(
            enc_Wih, enc_Whh, enc_bih, enc_bhh,
            dec_Wih, dec_Whh, dec_bih, dec_bhh,
            fc_W, enc_emb, dec_emb,
            encW, decW, encb, decb, fcWb, encEmB, decEmB,
            hbuf0, cbuf, tok);
    }

    bf16_t* hc = hbuf0;
    bf16_t* hn = hbuf1;

    // ---- encoder: 32 sequential fused embed+LSTM steps ----
    for (int t = 0; t < Ss; ++t) {
        k_lstm_step<<<dim3(Bb / 32, Hh / 64), 256, 0, stream>>>(
            encEmB, src, Ss, t, hc, hn, cbuf, encW, encb);
        bf16_t* tmp = hc; hc = hn; hn = tmp;
    }

    // ---- decoder: 48 sequential (fused embed+LSTM -> logits+argmax) steps ----
    for (int t = 0; t < Tt; ++t) {
        k_lstm_step<<<dim3(Bb / 32, Hh / 64), 256, 0, stream>>>(
            decEmB, tok, 1, 0, hc, hn, cbuf, decW, decb);
        { bf16_t* tmp = hc; hc = hn; hn = tmp; }
        k_logits_argmax<<<Bb / 16, 256, 0, stream>>>(hc, fcWb, fc_b, outp, tok, t);
    }
}
